// DCMHSA_30081950941548
// MI455X (gfx1250) — matrix operand order repaired and checked by reading, not compiled, not run on hardware
//
#include <hip/hip_runtime.h>
#include <hip/hip_bf16.h>

typedef __attribute__((ext_vector_type(16))) _Float16 v16h;
typedef __attribute__((ext_vector_type(8)))  _Float16 v8h;
typedef __attribute__((ext_vector_type(8)))  float    v8f;

#define BATCH 8
#define DIMC  384
#define OCH   1152
#define SPC   4096
#define NHEAD 8
#define DH    48
#define D2    24
#define DU    32

// ---------------- K0: cast x and w_qkv to f16 ----------------
__global__ void dcm_cast_k(const float* __restrict__ x, const float* __restrict__ w,
                           _Float16* __restrict__ xh, _Float16* __restrict__ wh) {
    long gid = (long)blockIdx.x * blockDim.x + threadIdx.x;
    const long NX = (long)BATCH * DIMC * SPC;        // 12,582,912
    const long NW = (long)OCH * DIMC;                // 442,368
    if (gid < NX)            xh[gid]      = (_Float16)x[gid];
    else if (gid < NX + NW)  wh[gid - NX] = (_Float16)w[gid - NX];
}

// ---------------- K1: qkv GEMM via WMMA f16 ----------------
// Per batch: Y[1152 x 4096] = W[1152 x 384] @ X[384 x 4096], f32 accum, f16 out.
// One wave computes a 16(M) x 64(N) strip.
__global__ __launch_bounds__(256) void dcm_qkv_gemm_k(const _Float16* __restrict__ Wh,
                                                      const _Float16* __restrict__ Xh,
                                                      _Float16* __restrict__ Yh) {
    int widb = threadIdx.x >> 5;
    int lane = threadIdx.x & 31;
    long wave = (long)blockIdx.x * 8 + widb;        // total 8*72*64 = 36864 waves
    int nstrip = (int)(wave % 64);
    int mtile  = (int)((wave / 64) % 72);
    int b      = (int)(wave / (64 * 72));
    const _Float16* X = Xh + (size_t)b * DIMC * SPC;
    _Float16*       Y = Yh + (size_t)b * OCH * SPC;
    int m0 = mtile * 16;
    int n0 = nstrip * 64;
    int r     = lane & 15;            // A row / C column within tile
    int khalf = (lane >> 4) * 8;      // lane-group K offset for A fragment
    v8f acc0 = {}, acc1 = {}, acc2 = {}, acc3 = {};
#pragma unroll 2
    for (int k0 = 0; k0 < DIMC; k0 += 32) {
        // ---- A fragment: lane holds row (m0+r), K = k0+khalf..+7 and k0+16+khalf..+7
        const _Float16* arow = Wh + (size_t)(m0 + r) * DIMC + k0 + khalf;
        v8h a_lo = *(const v8h*)(arow);
        v8h a_hi = *(const v8h*)(arow + 16);
        v16h a;
#pragma unroll
        for (int i = 0; i < 8; ++i) { a[i] = a_lo[i]; a[8 + i] = a_hi[i]; }
        // ---- B fragments: lane holds column n0+16j+r, K = k0+khalf..+7 and k0+16+khalf..+7
        const _Float16* bcol = X + (size_t)(k0 + khalf) * SPC + n0 + r;
        v16h b0, b1, b2, b3;
#pragma unroll
        for (int i = 0; i < 8; ++i) {
            const _Float16* p0 = bcol + (size_t)i * SPC;
            const _Float16* p1 = bcol + (size_t)(16 + i) * SPC;
            b0[i] = p0[0];  b0[8 + i] = p1[0];
            b1[i] = p0[16]; b1[8 + i] = p1[16];
            b2[i] = p0[32]; b2[8 + i] = p1[32];
            b3[i] = p0[48]; b3[8 + i] = p1[48];
        }
        acc0 = __builtin_amdgcn_wmma_f32_16x16x32_f16(false, a, false, b0, (short)0, acc0, false, false);
        acc1 = __builtin_amdgcn_wmma_f32_16x16x32_f16(false, a, false, b1, (short)0, acc1, false, false);
        acc2 = __builtin_amdgcn_wmma_f32_16x16x32_f16(false, a, false, b2, (short)0, acc2, false, false);
        acc3 = __builtin_amdgcn_wmma_f32_16x16x32_f16(false, a, false, b3, (short)0, acc3, false, false);
    }
    // C layout: lane col N = r; VGPR i -> row m0 + i + (lane>=16 ? 8 : 0)
    int mbase = m0 + ((lane >> 4) << 3);
#pragma unroll
    for (int i = 0; i < 8; ++i) {
        size_t rowoff = (size_t)(mbase + i) * SPC + n0 + r;
        Y[rowoff +  0] = (_Float16)acc0[i];
        Y[rowoff + 16] = (_Float16)acc1[i];
        Y[rowoff + 32] = (_Float16)acc2[i];
        Y[rowoff + 48] = (_Float16)acc3[i];
    }
}

// ---------------- K2: depthwise 3x3 conv, SAME padding ----------------
__global__ void dcm_dwconv_k(const _Float16* __restrict__ in, const float* __restrict__ w,
                             _Float16* __restrict__ out) {
    long gid = (long)blockIdx.x * blockDim.x + threadIdx.x;
    if (gid >= (long)BATCH * OCH * SPC) return;
    int  s  = (int)(gid & 4095);
    long cb = gid >> 12;                 // b*1152 + oc
    int  oc = (int)(cb % OCH);
    int x0 = s & 63, y0 = s >> 6;
    const _Float16* base = in + (cb << 12);
    const float* wk = w + oc * 9;
    float acc = 0.f;
#pragma unroll
    for (int ky = 0; ky < 3; ++ky) {
        int y = y0 + ky - 1;
        if ((unsigned)y >= 64u) continue;
#pragma unroll
        for (int kx = 0; kx < 3; ++kx) {
            int x = x0 + kx - 1;
            if ((unsigned)x >= 64u) continue;
            acc += (float)base[(y << 6) + x] * wk[ky * 3 + kx];
        }
    }
    out[gid] = (_Float16)acc;
}

// ---------------- K3: per-(b,head,d) row stats: scale, max, Z ----------------
__global__ __launch_bounds__(256) void dcm_rowstats_k(const _Float16* __restrict__ dwh,
                                                      const float* __restrict__ temperature,
                                                      float* __restrict__ stats) {
    int row = blockIdx.x;                 // 0..3071 = b*384 + hd
    int b = row / DIMC, hd = row % DIMC;
    int head = hd / DH;
    const _Float16* q = dwh + ((size_t)(b * OCH + hd) << 12);
    const _Float16* k = dwh + ((size_t)(b * OCH + DIMC + hd) << 12);
    int tid = threadIdx.x;
    float sq = 0.f, sk = 0.f, mx = -1e30f, mn = 1e30f;
    for (int s = tid; s < SPC; s += 256) {
        float qv = (float)q[s], kv = (float)k[s];
        sq += qv * qv; sk += kv * kv;
        float p = qv * kv;
        mx = fmaxf(mx, p); mn = fminf(mn, p);
    }
    __shared__ float s0[256], s1[256], s2[256], s3[256];
    s0[tid] = sq; s1[tid] = sk; s2[tid] = mx; s3[tid] = mn;
    for (int off = 128; off > 0; off >>= 1) {
        __syncthreads();
        if (tid < off) {
            s0[tid] += s0[tid + off]; s1[tid] += s1[tid + off];
            s2[tid] = fmaxf(s2[tid], s2[tid + off]);
            s3[tid] = fminf(s3[tid], s3[tid + off]);
        }
    }
    __syncthreads();
    __shared__ float sScale, sM;
    if (tid == 0) {
        float nq = fmaxf(sqrtf(s0[0]), 1e-12f);
        float nk = fmaxf(sqrtf(s1[0]), 1e-12f);
        float sc = temperature[head] / (nq * nk);
        sScale = sc;
        sM = (sc >= 0.f) ? sc * s2[0] : sc * s3[0];
    }
    __syncthreads();
    float sc = sScale, m = sM, z = 0.f;
    for (int s = tid; s < SPC; s += 256)
        z += __expf((float)q[s] * (float)k[s] * sc - m);
    __syncthreads();
    s0[tid] = z;
    for (int off = 128; off > 0; off >>= 1) {
        __syncthreads();
        if (tid < off) s0[tid] += s0[tid + off];
    }
    __syncthreads();
    if (tid == 0) {
        stats[row * 4 + 0] = sc; stats[row * 4 + 1] = m;
        stats[row * 4 + 2] = s0[0]; stats[row * 4 + 3] = 0.f;
    }
}

// ---------------- K4: fused cm-logit + ctx2 ----------------
// cml[b,h,s] = sum_d w_attn_r[d]*attn[d,s] ; ctx2[b,h,s] = (avg^T w_v_l) . v[:,s]
__global__ __launch_bounds__(256) void dcm_cml_ctx2_k(const _Float16* __restrict__ dwh,
        const float* __restrict__ stats, const float* __restrict__ w_attn_r,
        const float* __restrict__ w_attn_l, const float* __restrict__ w_v_l,
        float* __restrict__ cml, float* __restrict__ ctx2) {
    int blk = blockIdx.x;                 // 1024 blocks: (b, head, schunk)
    int schunk = blk & 15;
    int head = (blk >> 4) & 7;
    int b = blk >> 7;
    int tid = threadIdx.x;
    int s = schunk * 256 + tid;
    __shared__ float t24[24], wv[48], war[48], sSc[48], sMm[48], sZz[48];
    __shared__ float mxz[2];
    if (tid < 24) {
        float acc = 0.f;
        for (int d = 0; d < DH; ++d) acc += w_attn_l[tid * DH + d];
        t24[tid] = acc * (1.0f / 4096.0f);
    }
    if (tid < 48) {
        war[tid] = w_attn_r[tid];
        int row = b * DIMC + head * DH + tid;
        sSc[tid] = stats[row * 4 + 0];
        sMm[tid] = stats[row * 4 + 1];
        sZz[tid] = stats[row * 4 + 2];
    }
    __syncthreads();
    if (tid == 0) {
        float mx = t24[0];
        for (int o = 1; o < D2; ++o) mx = fmaxf(mx, t24[o]);
        float Z = 0.f;
        for (int o = 0; o < D2; ++o) Z += __expf(t24[o] - mx);
        mxz[0] = mx; mxz[1] = Z;
    }
    __syncthreads();
    if (tid < 48) {
        float acc = 0.f;
        for (int o = 0; o < D2; ++o)
            acc += (__expf(t24[o] - mxz[0]) / mxz[1]) * w_v_l[o * DH + tid];
        wv[tid] = acc;
    }
    __syncthreads();
    const _Float16* qb = dwh + ((size_t)(b * OCH + head * DH) << 12) + s;
    const _Float16* kb = qb + ((size_t)DIMC << 12);
    const _Float16* vb = qb + ((size_t)(2 * DIMC) << 12);
    float cacc = 0.f, c2 = 0.f;
    for (int d = 0; d < DH; ++d) {
        float qv = (float)qb[(size_t)d << 12];
        float kv = (float)kb[(size_t)d << 12];
        float vv = (float)vb[(size_t)d << 12];
        float a = __expf(qv * kv * sSc[d] - sMm[d]) / sZz[d];
        cacc += war[d] * a;
        c2   += wv[d] * vv;
    }
    int idx = ((b * NHEAD + head) << 12) + s;
    cml[idx]  = cacc;
    ctx2[idx] = c2;
}

// ---------------- K5: softmax(cm) over s, vbar = sum_s cm[s]*v[:,s], ctx = w_v_r @ vbar ---
__global__ __launch_bounds__(256) void dcm_ctx_k(const float* __restrict__ cml,
        const _Float16* __restrict__ dwh, const float* __restrict__ w_v_r,
        float* __restrict__ ctx) {
    int bh = blockIdx.x;                  // 64 blocks
    int b = bh >> 3, head = bh & 7;
    int tid = threadIdx.x;
    const float* L = cml + ((size_t)bh << 12);
    __shared__ float red[256];
    __shared__ float sMx, sZ, vbar[48];
    float mx = -1e30f;
    for (int s = tid; s < SPC; s += 256) mx = fmaxf(mx, L[s]);
    red[tid] = mx;
    for (int off = 128; off > 0; off >>= 1) {
        __syncthreads();
        if (tid < off) red[tid] = fmaxf(red[tid], red[tid + off]);
    }
    __syncthreads();
    if (tid == 0) sMx = red[0];
    __syncthreads();
    float zz = 0.f;
    for (int s = tid; s < SPC; s += 256) zz += __expf(L[s] - sMx);
    __syncthreads();
    red[tid] = zz;
    for (int off = 128; off > 0; off >>= 1) {
        __syncthreads();
        if (tid < off) red[tid] += red[tid + off];
    }
    __syncthreads();
    if (tid == 0) sZ = red[0];
    __syncthreads();
    const _Float16* vbase = dwh + ((size_t)(b * OCH + 2 * DIMC + head * DH) << 12);
    for (int d = 0; d < DH; ++d) {
        const _Float16* vr = vbase + ((size_t)d << 12);
        float acc = 0.f;
        for (int s = tid; s < SPC; s += 256)
            acc += (__expf(L[s] - sMx) / sZ) * (float)vr[s];
        __syncthreads();
        red[tid] = acc;
        for (int off = 128; off > 0; off >>= 1) {
            __syncthreads();
            if (tid < off) red[tid] += red[tid + off];
        }
        __syncthreads();
        if (tid == 0) vbar[d] = red[0];
    }
    __syncthreads();
    if (tid < D2) {
        float acc = 0.f;
        for (int d = 0; d < DH; ++d) acc += w_v_r[tid * DH + d] * vbar[d];
        ctx[bh * D2 + tid] = acc;
    }
}

// ---------------- K6: MLP + LayerNorm -> mask_ch[b, 384] (ch = d*8 + head) -------------
__global__ __launch_bounds__(256) void dcm_mlp_k(const float* __restrict__ ctx,
        const float* __restrict__ w_up1, const float* __restrict__ b_up1,
        const float* __restrict__ ln_w, const float* __restrict__ ln_b,
        const float* __restrict__ w_up2, const float* __restrict__ b_up2,
        float* __restrict__ mask_ch) {
    int b = blockIdx.x;
    int tid = threadIdx.x;
    __shared__ float sc[NHEAD][D2];
    __shared__ float red[256];
    __shared__ float sMu, sVar;
    __shared__ float r2[DU][NHEAD];
    if (tid < NHEAD * D2) sc[tid / D2][tid % D2] = ctx[b * NHEAD * D2 + tid];
    __syncthreads();
    int o = tid >> 3, h = tid & 7;       // o in [0,32), h in [0,8)
    float u = b_up1[o];
    for (int c = 0; c < D2; ++c) u += w_up1[o * D2 + c] * sc[h][c];
    red[tid] = u;
    for (int off = 128; off > 0; off >>= 1) {
        __syncthreads();
        if (tid < off) red[tid] += red[tid + off];
    }
    __syncthreads();
    if (tid == 0) sMu = red[0] / 256.f;
    __syncthreads();
    float dmu = u - sMu;
    red[tid] = dmu * dmu;
    for (int off = 128; off > 0; off >>= 1) {
        __syncthreads();
        if (tid < off) red[tid] += red[tid + off];
    }
    __syncthreads();
    if (tid == 0) sVar = red[0] / 256.f;
    __syncthreads();
    float un = dmu * rsqrtf(sVar + 1e-5f) * ln_w[o * NHEAD + h] + ln_b[o * NHEAD + h];
    r2[o][h] = fmaxf(un, 0.f);
    __syncthreads();
    for (int idx = tid; idx < DIMC; idx += 256) {
        int d = idx >> 3, hh = idx & 7;
        float v = b_up2[d];
        for (int oo = 0; oo < DU; ++oo) v += w_up2[d * DU + oo] * r2[oo][hh];
        mask_ch[b * DIMC + idx] = 1.f / (1.f + __expf(-v));
    }
}

// ---------------- K7: out = x * (w_proj @ sigmoid(ctx2) + mask_ch) ----------------
__global__ void dcm_final_k(const float* __restrict__ x, const float* __restrict__ ctx2,
        const float* __restrict__ w_proj, const float* __restrict__ mask_ch,
        float* __restrict__ out) {
    long gid = (long)blockIdx.x * blockDim.x + threadIdx.x;
    if (gid >= (long)BATCH * DIMC * SPC) return;
    int s  = (int)(gid & 4095);
    long cb = gid >> 12;
    int ch = (int)(cb % DIMC);
    int b  = (int)(cb / DIMC);
    const float* c2 = ctx2 + (((size_t)b * NHEAD) << 12) + s;
    float msp = 0.f;
#pragma unroll
    for (int h = 0; h < NHEAD; ++h) {
        float sg = 1.f / (1.f + __expf(-c2[(size_t)h << 12]));
        msp += w_proj[ch * NHEAD + h] * sg;
    }
    out[gid] = x[gid] * (msp + mask_ch[b * DIMC + ch]);
}

extern "C" void kernel_launch(void* const* d_in, const int* in_sizes, int n_in,
                              void* d_out, int out_size, void* d_ws, size_t ws_size,
                              hipStream_t stream) {
    const float* x       = (const float*)d_in[0];
    const float* temp    = (const float*)d_in[1];
    const float* w_qkv   = (const float*)d_in[2];
    const float* w_dw    = (const float*)d_in[3];
    const float* w_proj  = (const float*)d_in[4];
    const float* w_attnr = (const float*)d_in[5];
    const float* w_v_r   = (const float*)d_in[6];
    const float* w_up1   = (const float*)d_in[7];
    const float* b_up1   = (const float*)d_in[8];
    const float* ln_w    = (const float*)d_in[9];
    const float* ln_b    = (const float*)d_in[10];
    const float* w_up2   = (const float*)d_in[11];
    const float* b_up2   = (const float*)d_in[12];
    const float* w_attnl = (const float*)d_in[13];
    const float* w_v_l   = (const float*)d_in[14];
    float* out = (float*)d_out;

    char* ws = (char*)d_ws;
    const size_t SZ_XH   = (size_t)BATCH * DIMC * SPC * 2;   // 25,165,824
    const size_t SZ_WH   = (size_t)OCH * DIMC * 2;           //    884,736
    const size_t SZ_QKVH = (size_t)BATCH * OCH * SPC * 2;    // 75,497,472
    const size_t SZ_DWH  = SZ_QKVH;
    const size_t SZ_ST   = (size_t)BATCH * DIMC * 4 * 4;     //     49,152
    const size_t SZ_CML  = (size_t)BATCH * NHEAD * SPC * 4;  //  1,048,576
    const size_t SZ_CTX2 = SZ_CML;
    const size_t SZ_CTX  = (size_t)BATCH * NHEAD * D2 * 4;
    _Float16* xh     = (_Float16*)(ws);
    _Float16* wh     = (_Float16*)(ws + SZ_XH);
    _Float16* qkvh   = (_Float16*)(ws + SZ_XH + SZ_WH);
    _Float16* dwh    = (_Float16*)(ws + SZ_XH + SZ_WH + SZ_QKVH);
    float*    stats  = (float*)   (ws + SZ_XH + SZ_WH + SZ_QKVH + SZ_DWH);
    float*    cml    = (float*)   (ws + SZ_XH + SZ_WH + SZ_QKVH + SZ_DWH + SZ_ST);
    float*    ctx2   = (float*)   (ws + SZ_XH + SZ_WH + SZ_QKVH + SZ_DWH + SZ_ST + SZ_CML);
    float*    ctx    = (float*)   (ws + SZ_XH + SZ_WH + SZ_QKVH + SZ_DWH + SZ_ST + SZ_CML + SZ_CTX2);
    float*    maskch = (float*)   (ws + SZ_XH + SZ_WH + SZ_QKVH + SZ_DWH + SZ_ST + SZ_CML + SZ_CTX2 + SZ_CTX);

    // K0: cast
    {
        long n = (long)BATCH * DIMC * SPC + (long)OCH * DIMC;
        int blocks = (int)((n + 255) / 256);
        dcm_cast_k<<<blocks, 256, 0, stream>>>(x, w_qkv, xh, wh);
    }
    // K1: WMMA GEMM (8 waves/block; 36864 waves total)
    dcm_qkv_gemm_k<<<4608, 256, 0, stream>>>(wh, xh, qkvh);
    // K2: depthwise 3x3
    {
        long n = (long)BATCH * OCH * SPC;
        dcm_dwconv_k<<<(int)((n + 255) / 256), 256, 0, stream>>>(qkvh, w_dw, dwh);
    }
    // K3: row stats
    dcm_rowstats_k<<<BATCH * DIMC, 256, 0, stream>>>(dwh, temp, stats);
    // K4: fused cm-logit + ctx2
    dcm_cml_ctx2_k<<<BATCH * NHEAD * 16, 256, 0, stream>>>(dwh, stats, w_attnr, w_attnl, w_v_l, cml, ctx2);
    // K5: cm softmax + vbar + ctx
    dcm_ctx_k<<<BATCH * NHEAD, 256, 0, stream>>>(cml, dwh, w_v_r, ctx);
    // K6: MLP + LN -> channel mask
    dcm_mlp_k<<<BATCH, 256, 0, stream>>>(ctx, w_up1, b_up1, ln_w, ln_b, w_up2, b_up2, maskch);
    // K7: fused output
    {
        long n = (long)BATCH * DIMC * SPC;
        dcm_final_k<<<(int)((n + 255) / 256), 256, 0, stream>>>(x, ctx2, w_proj, maskch, out);
    }
}
